// LSTMnet_83124797047362
// MI455X (gfx1250) — compile-verified, matrix operand order checked by reading, not run on hardware
//
#include <hip/hip_runtime.h>
#include <math.h>

// ---------------------------------------------------------------------------
// CDNA5 / gfx1250 LSTM stack, v3.
//   - bf16 WMMA 16x16x32, f32 accumulation everywhere.
//   - xw (input projection + biases) injected via wmma(Identity, xwB, 0).
//   - Recurrent loop: double-buffered per-tile weight/xw register preload,
//     one-tile software pipeline -> loads clause-issue while the previous
//     tile's WMMA chain runs (v2 serialized load->wait->wmma per kc).
//   - Anti-LICM via an asm-touched *integer offset* (keeps global addrspace,
//     v2's pointer touch degraded weight loads to flat_load + dscnt waits).
//   - Gates fully in registers, c in VGPRs, h via 8KB LDS; fast tanh.
// ---------------------------------------------------------------------------

typedef __attribute__((ext_vector_type(16))) __bf16 v16bf;
typedef __attribute__((ext_vector_type(8)))  float  v8f;

constexpr int Bsz   = 256;
constexpr int Tlen  = 512;
constexpr int INDIM = 40;
constexpr int H1    = 256;
constexpr int G1    = 4 * H1;     // 1024 gate columns
constexpr int NT1   = G1 / 16;    // 64 N-tiles
constexpr int KCH1  = H1 / 32;    // 8 K-chunks of 32
constexpr int K0    = 64;         // input dim padded 40 -> 64
constexpr int KCH0  = K0 / 32;    // 2
constexpr int H2    = 8;
constexpr int G2    = 4 * H2;     // 32
constexpr int NT2   = G2 / 16;    // 2
constexpr int MT    = (Bsz * Tlen) / 16;  // 8192 row tiles (rows r' = t*B + b)
constexpr int TILE_E = 32 * 16;   // elements per packed operand tile (512)

__device__ __forceinline__ __bf16 f2bf(float f) {
  union { float f; unsigned u; } v; v.f = f;
  unsigned r = v.u + 0x7FFFu + ((v.u >> 16) & 1u);   // round-to-nearest-even
  unsigned short h = (unsigned short)(r >> 16);
  return __builtin_bit_cast(__bf16, h);
}
__device__ __forceinline__ float bf2f(__bf16 b) {
  unsigned short h = __builtin_bit_cast(unsigned short, b);
  union { unsigned u; float f; } v; v.u = ((unsigned)h) << 16;
  return v.f;
}
__device__ __forceinline__ __bf16 bf_zero() {
  return __builtin_bit_cast(__bf16, (unsigned short)0);
}
__device__ __forceinline__ float sigm(float x) { return 1.f / (1.f + __expf(-x)); }
__device__ __forceinline__ float tanh_fast(float x) {
  return 2.f / (1.f + __expf(-2.f * x)) - 1.f;   // 2*sigmoid(2x)-1
}

// --- weight repack: [4H][K] f32 row-major -> per-lane WMMA B layout, bf16 ---
// packed[nt][kc][lane][e] , n = nt*16 + lane%16 , k = kc*32 + (lane/16)*16 + e
__global__ void pack_w(const float* __restrict__ w, __bf16* __restrict__ wp,
                       int ntiles, int kch, int Nsrc, int Ksrc) {
  int idx = blockIdx.x * blockDim.x + threadIdx.x;
  int total = ntiles * kch * 32 * 16;
  if (idx >= total) return;
  int e    = idx & 15;
  int lane = (idx >> 4) & 31;
  int blk  = idx >> 9;            // nt*kch + kc
  int kc   = blk % kch;
  int nt   = blk / kch;
  int n = nt * 16 + (lane & 15);
  int k = kc * 32 + (lane >> 4) * 16 + e;
  float v = (n < Nsrc && k < Ksrc) ? w[n * Ksrc + k] : 0.f;
  wp[idx] = f2bf(v);
}

__global__ void bias_sum(const float* __restrict__ bi, const float* __restrict__ bh,
                         float* __restrict__ o, int n) {
  int i = blockIdx.x * blockDim.x + threadIdx.x;
  if (i < n) o[i] = bi[i] + bh[i];
}

// --- x (f32 [B][T][40]) -> bf16 [B][T][64] zero-padded -----------------------
__global__ void x_pad_bf16(const float* __restrict__ x, __bf16* __restrict__ act,
                           int nrows) {
  int idx = blockIdx.x * blockDim.x + threadIdx.x;
  if (idx >= nrows * K0) return;
  int k = idx % K0, r = idx / K0;
  float v = (k < INDIM) ? x[r * INDIM + k] : 0.f;
  act[idx] = f2bf(v);
}

// --- input-projection GEMM: xw[r', n] = act[b][t][:] . w_ih[n][:] + bias[n] --
// rows r' = t*B + b.  Output per tile in B-OPERAND layout ("xwB"):
//   tile[lane'][e] : lane' = column n (0..15) , e = row m  (k'=m slot),
//   lanes 16..31 (k'=16..31 slots) zeroed. The recurrent kernel injects the
//   tile via wmma(Identity, xwB, 0).
template <int KCH, int NTILES>
__global__ void xw_gemm(const __bf16* __restrict__ act,  // [B][T][KCH*32]
                        const __bf16* __restrict__ wp,   // packed weights
                        const float*  __restrict__ bias, // [NTILES*16]
                        __bf16* __restrict__ xwb) {
  const int mt    = blockIdx.x;
  const int lane  = threadIdx.x & 31;
  const int wave  = threadIdx.x >> 5;
  const int nwav  = blockDim.x >> 5;
  const int K     = KCH * 32;
  const int t     = mt >> 4;                       // B/16 = 16 tiles per step
  const int b     = ((mt & 15) << 4) + (lane & 15);
  const __bf16* arow = act + ((size_t)b * Tlen + t) * K + (lane >> 4) * 16;
  v16bf a[KCH];
#pragma unroll
  for (int kc = 0; kc < KCH; ++kc)
    a[kc] = *(const v16bf*)(arow + kc * 32);
  for (int nt = wave; nt < NTILES; nt += nwav) {
    const __bf16* wt = wp + ((size_t)(nt * KCH) * 32 + lane) * 16;
    v8f acc = {};
#pragma unroll
    for (int kc = 0; kc < KCH; ++kc) {
      v16bf wv = *(const v16bf*)(wt + (size_t)kc * TILE_E);
      acc = __builtin_amdgcn_wmma_f32_16x16x32_bf16(false, a[kc], false, wv,
                                                    (short)0, acc, false, false);
    }
    const float bc = bias[nt * 16 + (lane & 15)];
    // acc holds col n = lane&15, rows m = 8*(lane>>4)+v. Scatter into xwB:
    __bf16* tile = xwb + (size_t)(mt * NTILES + nt) * TILE_E;
    __bf16* o = tile + (lane & 15) * 16 + (lane >> 4) * 8;
#pragma unroll
    for (int v = 0; v < 8; ++v) o[v] = f2bf(acc[v] + bc);
    // zero the unused k'=16..31 operand half
    __bf16* z = tile + (16 + (lane & 15)) * 16 + (lane >> 4) * 8;
#pragma unroll
    for (int v = 0; v < 8; ++v) z[v] = bf_zero();
  }
}

// --- recurrent layer: one WG per 16 batch rows, all 512 steps in-kernel ------
// Wave w owns N-tiles {G*16 + 2w + j : G=0..3, j=0..1}  -> every lane holds
// i/f/g/o for the same (row,col); c lives in registers; h via 8KB LDS.
__global__ void __launch_bounds__(256, 1)
lstm_rec(const __bf16* __restrict__ whhp,  // packed [NT1][KCH1][32][16]
         const __bf16* __restrict__ xwb,   // packed xwB [MT][NT1][512]
         __bf16* __restrict__ actout) {    // [B][T][H1] bf16
  __shared__ __bf16 hS[16 * H1];           // 8 KB
  const int lane  = threadIdx.x & 31;
  const int wave  = threadIdx.x >> 5;
  const int bbase = blockIdx.x * 16;

  for (int i = threadIdx.x; i < 16 * H1; i += blockDim.x) hS[i] = bf_zero();

  // Identity A operand: A[m][k'] = (k'==m). Lane l<16 holds row m=l, slots
  // k'=0..15; lanes 16..31 (k'=16..31) are all zero.
  v16bf idA;
#pragma unroll
  for (int e = 0; e < 16; ++e)
    idA[e] = (lane < 16 && e == lane) ? f2bf(1.f) : bf_zero();

  float creg[2][8];
#pragma unroll
  for (int j = 0; j < 2; ++j)
#pragma unroll
    for (int v = 0; v < 8; ++v) creg[j][v] = 0.f;

  const int arow_off = (lane & 15) * H1 + (lane >> 4) * 16;
  const int rowv0    = (lane >> 4) * 8;            // first row this lane holds
  __syncthreads();

  for (int t = 0; t < Tlen; ++t) {
    // Anti-LICM: an opaque 0 offset makes the weight loads loop-variant so
    // they re-stream from (L2-resident) global instead of being hoisted into
    // 512 VGPRs and spilled. Integer (not pointer) touch keeps addrspace(1).
    size_t wofs = 0;
    asm volatile("" : "+s"(wofs));
    const __bf16* wbase = whhp + wofs;

    // A operand: previous h from LDS (coalesced b128 pairs)
    v16bf a[KCH1];
#pragma unroll
    for (int kc = 0; kc < KCH1; ++kc)
      a[kc] = *(const v16bf*)(hS + arow_off + kc * 32);
    __syncthreads();   // everyone has read old h before anyone overwrites it

    const int mt = t * 16 + blockIdx.x;

    // Double-buffered tile operands: loads for tile idx+1 clause-issue while
    // tile idx's 9-deep WMMA chain executes.
    v16bf wv[2][KCH1];
    v16bf xv[2];
    auto load_tile = [&](int slot, int idx) {
      const int nt = (idx >> 1) * 16 + (wave << 1) + (idx & 1);
      const __bf16* wt = wbase + ((size_t)(nt * KCH1) * 32 + lane) * 16;
#pragma unroll
      for (int kc = 0; kc < KCH1; ++kc)
        wv[slot][kc] = *(const v16bf*)(wt + (size_t)kc * TILE_E);
      const __bf16* xt = xwb + (size_t)(mt * NT1 + nt) * TILE_E;
      xv[slot] = *(const v16bf*)(xt + lane * 16);
      // prefetch next timestep's xw tile (16 tiles ahead in mt)
      __builtin_prefetch(xt + (size_t)16 * NT1 * TILE_E + lane * 16, 0, 3);
    };

    v8f acc[4][2];
    load_tile(0, 0);
#pragma unroll
    for (int idx = 0; idx < 8; ++idx) {
      if (idx < 7) load_tile((idx + 1) & 1, idx + 1);
      const int s = idx & 1;
      // inject xw (+biases) through the matrix pipe: c = I * xwB
      v8f c = __builtin_amdgcn_wmma_f32_16x16x32_bf16(false, idA, false, xv[s],
                                                      (short)0, (v8f){}, false, false);
#pragma unroll
      for (int kc = 0; kc < KCH1; ++kc)
        c = __builtin_amdgcn_wmma_f32_16x16x32_bf16(false, a[kc], false, wv[s][kc],
                                                    (short)0, c, false, false);
      acc[idx >> 1][idx & 1] = c;
    }

    // Gate math fully in registers; c state stays in VGPRs across steps.
#pragma unroll
    for (int j = 0; j < 2; ++j) {
      const int col = ((wave << 1) + j) * 16 + (lane & 15);
#pragma unroll
      for (int v = 0; v < 8; ++v) {
        float ig = sigm(acc[0][j][v]);
        float fg = sigm(acc[1][j][v]);
        float gg = tanh_fast(acc[2][j][v]);
        float og = sigm(acc[3][j][v]);
        float cc = fg * creg[j][v] + ig * gg;
        creg[j][v] = cc;
        float hh = og * tanh_fast(cc);
        hS[(rowv0 + v) * H1 + col] = f2bf(hh);
      }
    }
    __syncthreads();   // h tile complete

    // Coalesced h -> global activations (32B per thread)
    {
      const int rr = threadIdx.x >> 4;
      const int c0 = (threadIdx.x & 15) * 16;
      const __bf16* src = hS + rr * H1 + c0;
      __bf16* dst = actout + ((size_t)(bbase + rr) * Tlen + t) * H1 + c0;
      *(uint4*)(dst)     = *(const uint4*)(src);
      *(uint4*)(dst + 8) = *(const uint4*)(src + 8);
    }
  }
}

// --- final H=8 LSTM + FC head + log_softmax (1 thread per batch row) ---------
__global__ void lstm_small_head(const __bf16* __restrict__ xwb2, // xwB [MT][2][512]
                                const float* __restrict__ whh2,  // [32][8]
                                const float* __restrict__ fc_w,  // [4][8]
                                const float* __restrict__ fc_b,  // [4]
                                const float* __restrict__ fc2_w, // [2][4]
                                const float* __restrict__ fc2_b, // [2]
                                float* __restrict__ out) {       // [B][2]
  __shared__ float wS[G2 * H2];
  for (int i = threadIdx.x; i < G2 * H2; i += blockDim.x) wS[i] = whh2[i];
  __syncthreads();
  const int b = threadIdx.x;
  float h[8], c[8];
#pragma unroll
  for (int j = 0; j < 8; ++j) { h[j] = 0.f; c[j] = 0.f; }
  for (int t = 0; t < Tlen; ++t) {
    const int r = t * Bsz + b;
    const int mt = r >> 4, m = r & 15;
    float g[32];
#pragma unroll
    for (int n = 0; n < 32; ++n) {
      const int ntile = n >> 4, nn = n & 15;
      // xwB layout: tile[nn*16 + m]
      float gv = bf2f(xwb2[(size_t)(mt * NT2 + ntile) * TILE_E + nn * 16 + m]);
      const float* wr = wS + n * 8;
#pragma unroll
      for (int k = 0; k < 8; ++k) gv += h[k] * wr[k];
      g[n] = gv;
    }
#pragma unroll
    for (int j = 0; j < 8; ++j) {
      float ig = sigm(g[j]), fg = sigm(g[8 + j]);
      float gg = tanh_fast(g[16 + j]), og = sigm(g[24 + j]);
      c[j] = fg * c[j] + ig * gg;
      h[j] = og * tanh_fast(c[j]);
    }
  }
  float z[4];
#pragma unroll
  for (int i = 0; i < 4; ++i) {
    float s = fc_b[i];
#pragma unroll
    for (int k = 0; k < 8; ++k) s += fc_w[i * 8 + k] * h[k];
    z[i] = s;
  }
  float y[2];
#pragma unroll
  for (int i = 0; i < 2; ++i) {
    float s = fc2_b[i];
#pragma unroll
    for (int k = 0; k < 4; ++k) s += fc2_w[i * 4 + k] * z[k];
    y[i] = s;
  }
  float mx  = fmaxf(y[0], y[1]);
  float lse = mx + __logf(__expf(y[0] - mx) + __expf(y[1] - mx));
  out[b * 2 + 0] = y[0] - lse;
  out[b * 2 + 1] = y[1] - lse;
}

// ---------------------------------------------------------------------------
extern "C" void kernel_launch(void* const* d_in, const int* in_sizes, int n_in,
                              void* d_out, int out_size, void* d_ws, size_t ws_size,
                              hipStream_t stream) {
  (void)in_sizes; (void)n_in; (void)out_size; (void)ws_size;
  const float* x      = (const float*)d_in[0];
  const float* w_ih0  = (const float*)d_in[1];
  const float* w_hh0  = (const float*)d_in[2];
  const float* b_ih0  = (const float*)d_in[3];
  const float* b_hh0  = (const float*)d_in[4];
  const float* w_ih_r = (const float*)d_in[5];
  const float* w_hh_r = (const float*)d_in[6];
  const float* b_ih_r = (const float*)d_in[7];
  const float* b_hh_r = (const float*)d_in[8];
  const float* w_ih2  = (const float*)d_in[9];
  const float* w_hh2  = (const float*)d_in[10];
  const float* b_ih2  = (const float*)d_in[11];
  const float* b_hh2  = (const float*)d_in[12];
  const float* fc_w   = (const float*)d_in[13];
  const float* fc_b   = (const float*)d_in[14];
  const float* fc2_w  = (const float*)d_in[15];
  const float* fc2_b  = (const float*)d_in[16];
  float* out = (float*)d_out;

  // workspace carve-out
  char* base = (char*)d_ws;
  size_t off = 0;
  auto take = [&](size_t bytes) -> void* {
    void* p = base + off;
    off += (bytes + 255) & ~(size_t)255;
    return p;
  };
  const size_t wpk_sz = (size_t)NT1 * KCH1 * TILE_E * sizeof(__bf16);  // 512KB
  __bf16* wihp  = (__bf16*)take(wpk_sz);
  __bf16* whhp  = (__bf16*)take(wpk_sz);
  float*  biasb = (float*)take(G1 * sizeof(float));
  __bf16* actA  = (__bf16*)take((size_t)Bsz * Tlen * H1 * sizeof(__bf16)); // 64MB
  __bf16* actB  = (__bf16*)take((size_t)Bsz * Tlen * H1 * sizeof(__bf16)); // 64MB
  __bf16* xwb   = (__bf16*)take((size_t)MT * NT1 * TILE_E * sizeof(__bf16)); // 512MB

  // ---- layer 0: input 40 (padded 64) -> 256 ----
  {
    int tot = Bsz * Tlen * K0;
    x_pad_bf16<<<(tot + 255) / 256, 256, 0, stream>>>(x, actA, Bsz * Tlen);
    int pw0 = NT1 * KCH0 * 512;
    pack_w<<<(pw0 + 255) / 256, 256, 0, stream>>>(w_ih0, wihp, NT1, KCH0, G1, INDIM);
    int pw1 = NT1 * KCH1 * 512;
    pack_w<<<(pw1 + 255) / 256, 256, 0, stream>>>(w_hh0, whhp, NT1, KCH1, G1, H1);
    bias_sum<<<(G1 + 255) / 256, 256, 0, stream>>>(b_ih0, b_hh0, biasb, G1);
    xw_gemm<KCH0, NT1><<<MT, 256, 0, stream>>>(actA, wihp, biasb, xwb);
    lstm_rec<<<16, 256, 0, stream>>>(whhp, xwb, actB);
  }

  // ---- layers 1..4: 256 -> 256 ----
  __bf16* cur = actB;
  __bf16* nxt = actA;
  for (int l = 0; l < 4; ++l) {
    int pw = NT1 * KCH1 * 512;
    pack_w<<<(pw + 255) / 256, 256, 0, stream>>>(w_ih_r + (size_t)l * G1 * H1,
                                                 wihp, NT1, KCH1, G1, H1);
    pack_w<<<(pw + 255) / 256, 256, 0, stream>>>(w_hh_r + (size_t)l * G1 * H1,
                                                 whhp, NT1, KCH1, G1, H1);
    bias_sum<<<(G1 + 255) / 256, 256, 0, stream>>>(b_ih_r + (size_t)l * G1,
                                                   b_hh_r + (size_t)l * G1, biasb, G1);
    xw_gemm<KCH1, NT1><<<MT, 256, 0, stream>>>(cur, wihp, biasb, xwb);
    lstm_rec<<<16, 256, 0, stream>>>(whhp, xwb, nxt);
    __bf16* t2 = cur; cur = nxt; nxt = t2;
  }

  // ---- final layer: 256 -> 8, then FC head ----
  {
    int pw = NT2 * KCH1 * 512;
    pack_w<<<(pw + 255) / 256, 256, 0, stream>>>(w_ih2, wihp, NT2, KCH1, G2, H1);
    bias_sum<<<1, 256, 0, stream>>>(b_ih2, b_hh2, biasb, G2);
    xw_gemm<KCH1, NT2><<<MT, 64, 0, stream>>>(cur, wihp, biasb, xwb);
    lstm_small_head<<<1, 256, 0, stream>>>(xwb, w_hh2, fc_w, fc_b, fc2_w, fc2_b, out);
  }
}
